// TransformerBlock_21784074125884
// MI455X (gfx1250) — matrix operand order repaired and checked by reading, not compiled, not run on hardware
//
#include <hip/hip_runtime.h>
#include <hip/hip_bf16.h>

// ---------------------------------------------------------------------------
// MI455X (gfx1250) transformer block: bf16 WMMA flash-attention + FFN,
// with TDM (tensor_load_to_lds) double-buffered V staging.
// ---------------------------------------------------------------------------

typedef __attribute__((ext_vector_type(16))) __bf16       bf16x16;
typedef __attribute__((ext_vector_type(8)))  float        f32x8;
typedef __attribute__((ext_vector_type(4)))  unsigned int u32x4;
typedef __attribute__((ext_vector_type(4)))  int          i32x4;
typedef __attribute__((ext_vector_type(8)))  int          i32x8;

#define DIM   1024
#define NB    32
#define SEQ   1024
#define MTOT  (NB * SEQ)   // 32768 rows

union FragCvt { bf16x16 v; u32x4 q[2]; };

static __device__ __forceinline__ f32x8 wmma_bf16(bf16x16 a, bf16x16 b, f32x8 c) {
  // (neg_a, A, neg_b, B, c_mod, C, reuse_a, reuse_b)
  return __builtin_amdgcn_wmma_f32_16x16x32_bf16(false, a, false, b, (short)0, c,
                                                 false, false);
}

// A-matrix 16x32 bf16 fragment from a row-major source (global or LDS).
// Lane L<16: row=L, K=0..15; lane>=16: row=L-16, K=16..31.
static __device__ __forceinline__ bf16x16 load_fragA(const __bf16* p, int ld, int lane) {
  int r    = lane & 15;
  int koff = (lane >> 4) << 4;            // 0 or 16
  const __bf16* base = p + (size_t)r * ld + koff;
  FragCvt u;
  u.q[0] = *(const u32x4*)(base);         // K koff..koff+7
  u.q[1] = *(const u32x4*)(base + 8);     // K koff+8..koff+15
  return u.v;
}

// B-matrix 32x16 bf16 fragment where source rows are the N index (source is
// B^T, row-major, K along the row). Lane<16: K=0..15, lane>=16: K=16..31.
static __device__ __forceinline__ bf16x16 load_fragB_rows(const __bf16* p, int ld, int lane) {
  int n    = lane & 15;
  int koff = (lane >> 4) << 4;            // 0 or 16
  const __bf16* base = p + (size_t)n * ld + koff;
  FragCvt u;
  u.q[0] = *(const u32x4*)(base);
  u.q[1] = *(const u32x4*)(base + 8);
  return u.v;
}

// B-matrix 32x16 bf16 fragment from a K-major LDS tile: element [k][n] at
// p[k*ld + n]. Lanes vary N at fixed K -> bank-conflict-free.
static __device__ __forceinline__ bf16x16 load_fragB_cols(const __bf16* p, int ld, int lane) {
  int n    = lane & 15;
  int koff = (lane >> 4) << 4;
  bf16x16 f;
#pragma unroll
  for (int d = 0; d < 8; ++d) {
    int k = koff + 2 * d;
    f[2 * d]     = p[(size_t)k * ld + n];
    f[2 * d + 1] = p[(size_t)(k + 1) * ld + n];
  }
  return f;
}

// ---------------------------------------------------------------------------
// TDM: async-copy one contiguous 64KB tile (32768 bf16 elements) from global
// memory into LDS at byte offset lds_off. Descriptor per CDNA5 ISA ch.8:
//   group0: count=1 | lds_addr | global_addr (57b) | type=2
//   group1: data_size=1 (2B) | tensor_dim0=tile_dim0=32768 | dim1=1 | stride
// Tracked on TENSORcnt; completion published to other waves via barrier.
// ---------------------------------------------------------------------------
static __device__ __forceinline__ void tdm_load_64k(unsigned lds_off, const void* gsrc) {
  unsigned long long ga = (unsigned long long)gsrc;
  u32x4 g0;
  g0[0] = 1u;                                            // count=1, user desc
  g0[1] = lds_off;                                       // LDS byte address
  g0[2] = (unsigned)(ga & 0xFFFFFFFFu);                  // global_addr[31:0]
  g0[3] = (unsigned)((ga >> 32) & 0x01FFFFFFu) | (2u << 30);  // [56:32] | type=2
  i32x8 g1;
  g1[0] = 0x00010000;        // workgroup_mask=0, data_size=1 (2 bytes)
  g1[1] = (int)0x80000000;   // tensor_dim0[15:0]=32768 at bits 63:48
  g1[2] = 0x00010000;        // tensor_dim0 hi = 0; tensor_dim1 = 1
  g1[3] = (int)0x80000000;   // tile_dim0 = 32768 at bits 127:112
  g1[4] = 1;                 // tile_dim1 = 1
  g1[5] = 32768;             // tensor_dim0_stride (elements), low 32
  g1[6] = 0;
  g1[7] = 0;
  i32x4 z4; z4[0] = 0; z4[1] = 0; z4[2] = 0; z4[3] = 0;
#if defined(__clang_major__) && (__clang_major__ >= 23)
  i32x8 z8;
#pragma unroll
  for (int i = 0; i < 8; ++i) z8[i] = 0;
  __builtin_amdgcn_tensor_load_to_lds(g0, g1, z4, z4, z8, 0);
#else
  __builtin_amdgcn_tensor_load_to_lds(g0, g1, z4, z4, 0);
#endif
}

// ---------------------------------------------------------------------------
// fp32 -> bf16 bulk convert
// ---------------------------------------------------------------------------
__global__ void cvt_f32_bf16(const float* __restrict__ src, __bf16* __restrict__ dst, int n) {
  int i      = blockIdx.x * blockDim.x + threadIdx.x;
  int stride = gridDim.x * blockDim.x;
  for (int j = i * 4; j < n; j += stride * 4) {
    dst[j + 0] = (__bf16)src[j + 0];
    dst[j + 1] = (__bf16)src[j + 1];
    dst[j + 2] = (__bf16)src[j + 2];
    dst[j + 3] = (__bf16)src[j + 3];
  }
}

// ---------------------------------------------------------------------------
// Kernel 1: flash attention (16 q-rows / block) + residual + LayerNorm1.
// 8 waves; wave w owns output dims [128w, 128w+128).
// Dynamic LDS: V tiles (2 x 64KB, TDM double buffer) | S 16x34 f32 |
//              P 16x40 bf16 | m,l,alpha,mu,rs | LN partials 2x16x8 f32.
// ---------------------------------------------------------------------------
__launch_bounds__(256, 1)
__global__ void attn_ln1_kernel(const float*  __restrict__ Qf,
                                const __bf16* __restrict__ Qb,
                                const __bf16* __restrict__ Kb,
                                const __bf16* __restrict__ Vb,
                                const float*  __restrict__ mask,
                                const float*  __restrict__ g1,
                                const float*  __restrict__ be1,
                                float*  __restrict__ Xf,
                                __bf16* __restrict__ Xb) {
  extern __shared__ __align__(16) unsigned char smem[];
  // [0,64K) Vs buf0, [64K,128K) Vs buf1
  float*  Ss   = (float*)(smem + 131072);                     // 2176 B
  __bf16* Ps   = (__bf16*)(smem + 131072 + 2176);             // 1280 B
  float*  m_s  = (float*)(smem + 131072 + 2176 + 1280);
  float*  l_s  = m_s + 16;
  float*  al_s = l_s + 16;
  float*  mu_s = al_s + 16;
  float*  rs_s = mu_s + 16;
  float*  part = rs_s + 16;                                   // 256 floats

  const int tid  = threadIdx.x;
  const int lane = tid & 31;
  const int wave = tid >> 5;
  const int hi   = lane >> 4;
  const int b    = blockIdx.x >> 6;
  const int q0   = (blockIdx.x & 63) << 4;
  const int wslice = wave << 7;                               // 128 * wave
  const float scale = 1.0f / (32.0f + 1e-8f);                 // 1/(sqrt(1024)+eps)
  const unsigned lds_base = (unsigned)__builtin_amdgcn_groupstaticsize();

  if (tid < 16) { m_s[tid] = -3.0e38f; l_s[tid] = 0.0f; }

  // Persistent Q fragments for this wave's 128-dim slice (4 K-steps of 32).
  const __bf16* Qbase = Qb + (size_t)(b * SEQ + q0) * DIM + wslice;
  bf16x16 qf[4];
#pragma unroll
  for (int ks = 0; ks < 4; ++ks) qf[ks] = load_fragA(Qbase + ks * 32, DIM, lane);

  f32x8 acc[8];
#pragma unroll
  for (int f = 0; f < 8; ++f)
#pragma unroll
    for (int e = 0; e < 8; ++e) acc[f][e] = 0.0f;

  const __bf16* Kbatch   = Kb + (size_t)b * SEQ * DIM;
  const __bf16* Vbatch   = Vb + (size_t)b * SEQ * DIM;
  const float*  maskbase = mask + (size_t)(b * SEQ + q0) * SEQ;

  // Kick off TDM copy of V tile 0 into buffer 0.
  if (wave == 0) tdm_load_64k(lds_base, Vbatch);

  for (int it = 0; it < SEQ / 32; ++it) {
    const int cur = it & 1;
    const int k0  = it * 32;
    // Issue TDM for the NEXT tile into the spare buffer (its last readers
    // finished at the barrier that closed iteration it-1), then wait until
    // the CURRENT tile has landed (TDM ops complete in order per wave).
    if (wave == 0) {
      if (it + 1 < SEQ / 32) {
        tdm_load_64k(lds_base + (unsigned)((1 - cur) << 16),
                     Vbatch + (size_t)(it + 1) * 32 * DIM);
        __builtin_amdgcn_s_wait_tensorcnt(1);   // current tile landed
      } else {
        __builtin_amdgcn_s_wait_tensorcnt(0);   // last tile landed
      }
    }
    for (int i = tid; i < 16 * 34; i += 256) Ss[i] = 0.0f;
    __syncthreads();
    const __bf16* Vs = (const __bf16*)(smem + (cur << 16));

    // Partial S = Q[:,slice] * K_tile[:,slice]^T, combined via LDS f32 atomics.
#pragma unroll
    for (int nt = 0; nt < 2; ++nt) {
      f32x8 sacc;
#pragma unroll
      for (int e = 0; e < 8; ++e) sacc[e] = 0.0f;
#pragma unroll
      for (int ks = 0; ks < 4; ++ks) {
        bf16x16 kb = load_fragB_rows(
            Kbatch + (size_t)(k0 + nt * 16) * DIM + wslice + ks * 32, DIM, lane);
        sacc = wmma_bf16(qf[ks], kb, sacc);
      }
      // Warm GL2 with the next K tile rows for this slice.
      __builtin_prefetch(Kbatch + (size_t)(((k0 + 32) & (SEQ - 1)) + nt * 16) * DIM + wslice, 0, 1);
#pragma unroll
      for (int e = 0; e < 8; ++e) {
        int row = e + (hi << 3);
        atomicAdd(&Ss[row * 34 + nt * 16 + (lane & 15)], sacc[e]);
      }
    }
    __syncthreads();

    // Online softmax: one thread per q-row.
    if (tid < 16) {
      int r = tid;
      const float* mrow = maskbase + (size_t)r * SEQ + k0;
      float mmax = m_s[r];
      float sv[32];
#pragma unroll
      for (int j = 0; j < 32; ++j) {
        float s = Ss[r * 34 + j] * scale + mrow[j];
        sv[j]   = s;
        mmax    = fmaxf(mmax, s);
      }
      float a    = __expf(m_s[r] - mmax);
      float lnew = l_s[r] * a;
#pragma unroll
      for (int j = 0; j < 32; ++j) {
        float p = __expf(sv[j] - mmax);
        lnew += p;
        Ps[r * 40 + j] = (__bf16)p;
      }
      m_s[r] = mmax; l_s[r] = lnew; al_s[r] = a;
    }
    __syncthreads();

    // Rescale accumulators and add P * V_slice.
    {
      float alr[8];
#pragma unroll
      for (int e = 0; e < 8; ++e) alr[e] = al_s[e + (hi << 3)];
      bf16x16 pa = load_fragA(Ps, 40, lane);
#pragma unroll
      for (int f = 0; f < 8; ++f) {
#pragma unroll
        for (int e = 0; e < 8; ++e) acc[f][e] *= alr[e];
        bf16x16 vb = load_fragB_cols(Vs + wslice + f * 16, DIM, lane);
        acc[f] = wmma_bf16(pa, vb, acc[f]);
      }
    }
    __syncthreads();   // all waves done reading Vs[cur]; TDM may overwrite next iter
  }

  // Epilogue: o = acc/l + Q residual; LayerNorm1; write X (f32 + bf16).
  float lr[8];
#pragma unroll
  for (int e = 0; e < 8; ++e) lr[e] = 1.0f / l_s[e + (hi << 3)];
  const float* Qres = Qf + (size_t)(b * SEQ + q0) * DIM;
#pragma unroll
  for (int f = 0; f < 8; ++f) {
    int col = wslice + f * 16 + (lane & 15);
#pragma unroll
    for (int e = 0; e < 8; ++e) {
      int row = e + (hi << 3);
      acc[f][e] = acc[f][e] * lr[e] + Qres[(size_t)row * DIM + col];
    }
  }
  // Per-row partial sums (each half-wave covers rows hi*8..hi*8+7).
  float psum[8], psq[8];
#pragma unroll
  for (int e = 0; e < 8; ++e) { psum[e] = 0.0f; psq[e] = 0.0f; }
#pragma unroll
  for (int f = 0; f < 8; ++f)
#pragma unroll
    for (int e = 0; e < 8; ++e) { psum[e] += acc[f][e]; psq[e] += acc[f][e] * acc[f][e]; }
#pragma unroll
  for (int m = 1; m <= 8; m <<= 1) {
#pragma unroll
    for (int e = 0; e < 8; ++e) {
      psum[e] += __shfl_xor(psum[e], m);
      psq[e]  += __shfl_xor(psq[e],  m);
    }
  }
  if ((lane & 15) == 0) {
#pragma unroll
    for (int e = 0; e < 8; ++e) {
      int row = e + (hi << 3);
      part[row * 8 + wave]       = psum[e];
      part[128 + row * 8 + wave] = psq[e];
    }
  }
  __syncthreads();
  if (tid < 16) {
    float s = 0.0f, sq = 0.0f;
#pragma unroll
    for (int w = 0; w < 8; ++w) { s += part[tid * 8 + w]; sq += part[128 + tid * 8 + w]; }
    float mu  = s * (1.0f / 1024.0f);
    float var = sq * (1.0f / 1024.0f) - mu * mu;
    mu_s[tid] = mu;
    rs_s[tid] = rsqrtf(var + 1e-5f);
  }
  __syncthreads();
#pragma unroll
  for (int f = 0; f < 8; ++f) {
    int   col = wslice + f * 16 + (lane & 15);
    float g   = g1[col], bb = be1[col];
#pragma unroll
    for (int e = 0; e < 8; ++e) {
      int    row = e + (hi << 3);
      float  y   = (acc[f][e] - mu_s[row]) * rs_s[row] * g + bb;
      size_t gi  = (size_t)(b * SEQ + q0 + row) * DIM + col;
      Xf[gi] = y;
      Xb[gi] = (__bf16)y;
    }
  }
}

// ---------------------------------------------------------------------------
// Kernel 2: H = relu(X @ W1^T + b1). 64(M) x 128(N) block tile, 8 waves.
// ---------------------------------------------------------------------------
__launch_bounds__(256, 2)
__global__ void ffn1_kernel(const __bf16* __restrict__ Xb,
                            const __bf16* __restrict__ W1b,
                            const float*  __restrict__ b1,
                            __bf16* __restrict__ Hb) {
  __shared__ __align__(16) __bf16 As[64 * 40];
  __shared__ __align__(16) __bf16 Bs[128 * 40];
  const int tid  = threadIdx.x;
  const int lane = tid & 31;
  const int wave = tid >> 5;
  const int hi   = lane >> 4;
  const int m0   = (blockIdx.x >> 3) * 64;
  const int n0   = (blockIdx.x & 7) * 128;
  const int wm   = (wave & 1) * 32;
  const int wn   = (wave >> 1) * 32;

  f32x8 acc[2][2];
#pragma unroll
  for (int i = 0; i < 2; ++i)
#pragma unroll
    for (int j = 0; j < 2; ++j)
#pragma unroll
      for (int e = 0; e < 8; ++e) acc[i][j][e] = 0.0f;

  for (int k0 = 0; k0 < DIM; k0 += 32) {
    {
      int r = tid >> 2, c = tid & 3;   // 64 rows x 4 chunks
      *(u32x4*)&As[r * 40 + c * 8] =
          *(const u32x4*)(Xb + (size_t)(m0 + r) * DIM + k0 + c * 8);
    }
    for (int i = tid; i < 512; i += 256) {   // 128 rows x 4 chunks
      int r = i >> 2, c = i & 3;
      *(u32x4*)&Bs[r * 40 + c * 8] =
          *(const u32x4*)(W1b + (size_t)(n0 + r) * DIM + k0 + c * 8);
    }
    __syncthreads();
    bf16x16 af[2], bfr[2];
    af[0]  = load_fragA(As + (wm + 0) * 40, 40, lane);
    af[1]  = load_fragA(As + (wm + 16) * 40, 40, lane);
    bfr[0] = load_fragB_rows(Bs + (wn + 0) * 40, 40, lane);
    bfr[1] = load_fragB_rows(Bs + (wn + 16) * 40, 40, lane);
#pragma unroll
    for (int i = 0; i < 2; ++i)
#pragma unroll
      for (int j = 0; j < 2; ++j) acc[i][j] = wmma_bf16(af[i], bfr[j], acc[i][j]);
    __syncthreads();
  }

#pragma unroll
  for (int j = 0; j < 2; ++j) {
    int   col = n0 + wn + j * 16 + (lane & 15);
    float bb  = b1[col];
#pragma unroll
    for (int i = 0; i < 2; ++i)
#pragma unroll
      for (int e = 0; e < 8; ++e) {
        int   row = m0 + wm + i * 16 + e + (hi << 3);
        float h   = fmaxf(acc[i][j][e] + bb, 0.0f);
        Hb[(size_t)row * DIM + col] = (__bf16)h;
      }
  }
}

// ---------------------------------------------------------------------------
// Kernel 3: out = LN2(H @ W2^T + b2 + X). 16 rows/block over full D=1024.
// ---------------------------------------------------------------------------
__launch_bounds__(256, 1)
__global__ void ffn2_ln2_kernel(const __bf16* __restrict__ Hb,
                                const __bf16* __restrict__ W2b,
                                const float*  __restrict__ b2,
                                const float*  __restrict__ Xf,
                                const float*  __restrict__ g2,
                                const float*  __restrict__ be2,
                                float* __restrict__ out) {
  __shared__ __align__(16) __bf16 As[16 * 40];
  __shared__ float part[2 * 16 * 8];
  __shared__ float mu_s[16], rs_s[16];
  const int tid    = threadIdx.x;
  const int lane   = tid & 31;
  const int wave   = tid >> 5;
  const int hi     = lane >> 4;
  const int m0     = blockIdx.x * 16;
  const int wslice = wave << 7;

  f32x8 acc[8];
#pragma unroll
  for (int f = 0; f < 8; ++f)
#pragma unroll
    for (int e = 0; e < 8; ++e) acc[f][e] = 0.0f;

  for (int k0 = 0; k0 < DIM; k0 += 32) {
    if (tid < 64) {
      int r = tid >> 2, c = tid & 3;
      *(u32x4*)&As[r * 40 + c * 8] =
          *(const u32x4*)(Hb + (size_t)(m0 + r) * DIM + k0 + c * 8);
    }
    __syncthreads();
    bf16x16 a = load_fragA(As, 40, lane);
#pragma unroll
    for (int f = 0; f < 8; ++f) {
      bf16x16 w = load_fragB_rows(W2b + (size_t)(wslice + f * 16) * DIM + k0, DIM, lane);
      acc[f] = wmma_bf16(a, w, acc[f]);
    }
    __syncthreads();
  }

  // + b2 + residual X
#pragma unroll
  for (int f = 0; f < 8; ++f) {
    int   col = wslice + f * 16 + (lane & 15);
    float bb  = b2[col];
#pragma unroll
    for (int e = 0; e < 8; ++e) {
      int row = e + (hi << 3);
      acc[f][e] += bb + Xf[(size_t)(m0 + row) * DIM + col];
    }
  }

  // LayerNorm2 (register partials + half-wave shuffles + LDS reduce)
  float psum[8], psq[8];
#pragma unroll
  for (int e = 0; e < 8; ++e) { psum[e] = 0.0f; psq[e] = 0.0f; }
#pragma unroll
  for (int f = 0; f < 8; ++f)
#pragma unroll
    for (int e = 0; e < 8; ++e) { psum[e] += acc[f][e]; psq[e] += acc[f][e] * acc[f][e]; }
#pragma unroll
  for (int m = 1; m <= 8; m <<= 1) {
#pragma unroll
    for (int e = 0; e < 8; ++e) {
      psum[e] += __shfl_xor(psum[e], m);
      psq[e]  += __shfl_xor(psq[e],  m);
    }
  }
  if ((lane & 15) == 0) {
#pragma unroll
    for (int e = 0; e < 8; ++e) {
      int row = e + (hi << 3);
      part[row * 8 + wave]       = psum[e];
      part[128 + row * 8 + wave] = psq[e];
    }
  }
  __syncthreads();
  if (tid < 16) {
    float s = 0.0f, sq = 0.0f;
#pragma unroll
    for (int w = 0; w < 8; ++w) { s += part[tid * 8 + w]; sq += part[128 + tid * 8 + w]; }
    float mu  = s * (1.0f / 1024.0f);
    float var = sq * (1.0f / 1024.0f) - mu * mu;
    mu_s[tid] = mu;
    rs_s[tid] = rsqrtf(var + 1e-5f);
  }
  __syncthreads();
#pragma unroll
  for (int f = 0; f < 8; ++f) {
    int   col = wslice + f * 16 + (lane & 15);
    float g   = g2[col], bb = be2[col];
#pragma unroll
    for (int e = 0; e < 8; ++e) {
      int row = e + (hi << 3);
      out[(size_t)(m0 + row) * DIM + col] =
          (acc[f][e] - mu_s[row]) * rs_s[row] * g + bb;
    }
  }
}

// ---------------------------------------------------------------------------
// Host launcher. Workspace layout (bytes):
//   [0,64M)    Q bf16      [64M,128M)  K bf16     [128M,192M) V bf16
//   [192M,194M) W1 bf16    [194M,196M) W2 bf16
//   [196M,324M) X f32      [324M,388M) X bf16     [388M,452M) H bf16
// ---------------------------------------------------------------------------
extern "C" void kernel_launch(void* const* d_in, const int* in_sizes, int n_in,
                              void* d_out, int out_size, void* d_ws, size_t ws_size,
                              hipStream_t stream) {
  const float* Q    = (const float*)d_in[0];
  const float* K    = (const float*)d_in[1];
  const float* V    = (const float*)d_in[2];
  const float* mask = (const float*)d_in[3];
  const float* W1   = (const float*)d_in[4];
  const float* b1   = (const float*)d_in[5];
  const float* W2   = (const float*)d_in[6];
  const float* b2   = (const float*)d_in[7];
  const float* g1   = (const float*)d_in[8];
  const float* be1  = (const float*)d_in[9];
  const float* g2   = (const float*)d_in[10];
  const float* be2  = (const float*)d_in[11];
  float* out = (float*)d_out;

  const size_t MB = (size_t)1 << 20;
  char* ws = (char*)d_ws;
  __bf16* Qb  = (__bf16*)(ws + 0);
  __bf16* Kb  = (__bf16*)(ws + 64 * MB);
  __bf16* Vb  = (__bf16*)(ws + 128 * MB);
  __bf16* W1b = (__bf16*)(ws + 192 * MB);
  __bf16* W2b = (__bf16*)(ws + 194 * MB);
  float*  Xf  = (float*)(ws + 196 * MB);
  __bf16* Xb  = (__bf16*)(ws + 324 * MB);
  __bf16* Hb  = (__bf16*)(ws + 388 * MB);

  const int nQKV = MTOT * DIM;   // 33,554,432
  const int nW   = DIM * DIM;    // 1,048,576
  cvt_f32_bf16<<<2048, 256, 0, stream>>>(Q,  Qb,  nQKV);
  cvt_f32_bf16<<<2048, 256, 0, stream>>>(K,  Kb,  nQKV);
  cvt_f32_bf16<<<2048, 256, 0, stream>>>(V,  Vb,  nQKV);
  cvt_f32_bf16<<<256,  256, 0, stream>>>(W1, W1b, nW);
  cvt_f32_bf16<<<256,  256, 0, stream>>>(W2, W2b, nW);

  // dyn LDS: 2 x 64KB V buffers + S(2176) + P(1280) + stats(320) + part(1024)
  const int attn_smem = 131072 + 2176 + 1280 + 5 * 16 * 4 + 256 * 4;   // 135872
  attn_ln1_kernel<<<NB * (SEQ / 16), 256, attn_smem, stream>>>(
      Q, Qb, Kb, Vb, mask, g1, be1, Xf, Xb);

  ffn1_kernel<<<(MTOT / 64) * (DIM / 128), 256, 0, stream>>>(Xb, W1b, b1, Hb);

  ffn2_ln2_kernel<<<MTOT / 16, 256, 0, stream>>>(Hb, W2b, b2, Xf, g2, be2, out);
}
